// HSL_Layer_Part1_34368328303051
// MI455X (gfx1250) — matrix operand order repaired and checked by reading, not compiled, not run on hardware
//
#include <hip/hip_runtime.h>
#include <hip/hip_bf16.h>
#include <math.h>

#define NUM_NODES 512
#define NUM_EDGES 512
#define EMB_DIM   128
#define HIDDEN    256
#define NUM_INC   8192

typedef __attribute__((ext_vector_type(16))) _Float16 v16h;
typedef __attribute__((ext_vector_type(8)))  float    v8f;

// ---------------------------------------------------------------------------
// Phase 1: deterministic scatter-mean, one block per edge.
// Parallel match-compaction: 128 threads scan 64-entry chunks of E (staged in
// LDS), a prefix sum over per-chunk counts gives stable offsets, and matches
// are compacted in increasing-i order -> bitwise-deterministic accumulation,
// ~270 loop iterations instead of a serial 8192-scan. LDS: 64KB of 320KB/WGP.
// ---------------------------------------------------------------------------
__global__ void edge_mean_kernel(const float* __restrict__ X,
                                 const int*   __restrict__ V,
                                 const int*   __restrict__ E,
                                 float*       __restrict__ eX) {
  __shared__ int sE[NUM_INC];        // 32KB: edge ids
  __shared__ int sList[NUM_INC];     // 32KB: compacted matching node ids
  __shared__ int sCnt[129];
  const int tid = threadIdx.x;       // 0..127 (dim id / scanner id)
  const int e   = blockIdx.x;

  for (int i = tid; i < NUM_INC; i += 128) sE[i] = E[i];
  __syncthreads();

  const int CHUNK = NUM_INC / 128;   // 64
  const int c0 = tid * CHUNK;
  int cnt = 0;
  for (int j = 0; j < CHUNK; ++j) cnt += (sE[c0 + j] == e) ? 1 : 0;
  sCnt[tid + 1] = cnt;
  __syncthreads();
  if (tid == 0) {
    sCnt[0] = 0;
    for (int j = 1; j <= 128; ++j) sCnt[j] += sCnt[j - 1];
  }
  __syncthreads();
  int ofs = sCnt[tid];
  for (int j = 0; j < CHUNK; ++j)
    if (sE[c0 + j] == e) sList[ofs++] = V[c0 + j];
  __syncthreads();

  const int total = sCnt[128];
  float acc = 0.f;
  for (int j = 0; j < total; ++j)
    acc += X[sList[j] * EMB_DIM + tid];         // coalesced row reads
  eX[e * EMB_DIM + tid] = (total > 0) ? acc / (float)total : 0.f;
}

// ---------------------------------------------------------------------------
// Phase 2: WMMA GEMMs (template on which half of W1 / which layout).
//   ISE=false: hX [n,h] = sum_k X [n,k]*W1[h,k]       -> row-major [n][h]
//   ISE=true : hET[h,m] = sum_k eX[m,k]*W1[h,128+k]   -> transposed [h][m]
// One 16x16 tile per wave, K=128 fully unrolled into 4 v_wmma_f32_16x16x32_f16.
// A as loaded here (contiguous k, not ISA table order): lane l -> M=l&15, g=l>>4; e -> K=g*16+e.
//   B (32x16), same order as A: lane l -> N=l&15; e -> K=g*16+e.
// D (f32 16x16): lane l -> N=l&15; VGPR r -> M=(l>>4)*8+r.
// ---------------------------------------------------------------------------
template <bool ISE>
__global__ void gemm_wmma_kernel(const float* __restrict__ src,
                                 const float* __restrict__ W1,
                                 float*       __restrict__ dst) {
  const int wave = (blockIdx.x * blockDim.x + threadIdx.x) >> 5;   // 0..511
  const int lane = threadIdx.x & 31;
  const int n0   = (wave >> 4) << 4;   // row-tile base (n or m)
  const int h0   = (wave & 15) << 4;   // col-tile base (h)
  const int g    = lane >> 4;
  const int mm   = lane & 15;
  const int wofs = ISE ? EMB_DIM : 0;

  v8f c = {};
#pragma unroll
  for (int k0 = 0; k0 < EMB_DIM; k0 += 32) {
    v16h a, b;
#pragma unroll
    for (int ee = 0; ee < 8; ++ee) {
      a[ee]     = (_Float16)src[(n0 + mm) * EMB_DIM + k0 + g * 16 + ee];
      a[ee + 8] = (_Float16)src[(n0 + mm) * EMB_DIM + k0 + g * 16 + 8 + ee];
    }
#pragma unroll
    for (int ee = 0; ee < 16; ++ee) {
      b[ee] = (_Float16)W1[(h0 + mm) * (2 * EMB_DIM) + wofs + k0 + g * 16 + ee];
    }
    // (neg_a, A, neg_b, B, c_mod, C, reuse_a, reuse_b)
    c = __builtin_amdgcn_wmma_f32_16x16x32_f16(false, a, false, b,
                                               (short)0, c, false, false);
  }
#pragma unroll
  for (int r = 0; r < 8; ++r) {
    const int row = n0 + g * 8 + r;   // n (or m)
    const int col = h0 + mm;          // h
    if (ISE) dst[col * NUM_EDGES + row] = c[r];   // hET[h][m]
    else     dst[row * HIDDEN    + col] = c[r];   // hX [n][h]
  }
}

// ---------------------------------------------------------------------------
// Phase 3 (dominant): relu-reduce against W2 with an n-tile of NT rows per
// block. Each coalesced hET[h][m] load is reused NT times from registers,
// cutting L2 traffic NT-fold (hET itself is 512KB, fully L2-resident).
// ---------------------------------------------------------------------------
template <int NT>
__global__ void score_kernel(const float* __restrict__ hX,
                             const float* __restrict__ hET,
                             const float* __restrict__ b1,
                             const float* __restrict__ W2,
                             const float* __restrict__ b2,
                             float*       __restrict__ out) {
  __shared__ float sxb[NT][HIDDEN];
  __shared__ float sw2[HIDDEN];
  const int n0 = blockIdx.x * NT;
  const int m  = threadIdx.x;                   // 0..511
  for (int idx = m; idx < NT * HIDDEN; idx += NUM_EDGES) {
    const int nn = idx / HIDDEN, hh = idx % HIDDEN;
    sxb[nn][hh] = hX[(n0 + nn) * HIDDEN + hh] + b1[hh];
  }
  if (m < HIDDEN) sw2[m] = W2[m];
  __syncthreads();

  float acc[NT];
#pragma unroll
  for (int nn = 0; nn < NT; ++nn) acc[nn] = 0.f;

  for (int h = 0; h < HIDDEN; ++h) {
    if ((h & 31) == 0 && (h + 32) < HIDDEN)
      __builtin_prefetch(&hET[(h + 32) * NUM_EDGES + m], 0, 0);
    const float he = hET[h * NUM_EDGES + m];    // coalesced, reused NT times
    const float w  = sw2[h];                    // LDS broadcast
#pragma unroll
    for (int nn = 0; nn < NT; ++nn) {
      const float tv = fmaxf(sxb[nn][h] + he, 0.f);
      acc[nn] = fmaf(tv, w, acc[nn]);
    }
  }

  const float bias2 = b2[0];
#pragma unroll
  for (int nn = 0; nn < NT; ++nn) {
    float p = 1.0f / (1.0f + __expf(-(acc[nn] + bias2)));
    if (isnan(p)) p = 0.5f;
    p = fminf(fmaxf(p, 1e-6f), 1.0f - 1e-6f);
    out[(n0 + nn) * NUM_EDGES + m] = p;
  }
}

// ---------------------------------------------------------------------------
extern "C" void kernel_launch(void* const* d_in, const int* in_sizes, int n_in,
                              void* d_out, int out_size, void* d_ws, size_t ws_size,
                              hipStream_t stream) {
  const float* X  = (const float*)d_in[0];
  const int*   V  = (const int*)  d_in[1];
  const int*   E  = (const int*)  d_in[2];
  const float* W1 = (const float*)d_in[3];
  const float* b1 = (const float*)d_in[4];
  const float* W2 = (const float*)d_in[5];
  const float* b2 = (const float*)d_in[6];
  float* out = (float*)d_out;

  char* ws = (char*)d_ws;
  float* eX  = (float*)(ws);
  float* hX  = (float*)(ws + (size_t)NUM_EDGES * EMB_DIM * sizeof(float));
  float* hET = (float*)(ws + (size_t)NUM_EDGES * EMB_DIM * sizeof(float)
                           + (size_t)NUM_NODES * HIDDEN  * sizeof(float));

  edge_mean_kernel<<<NUM_EDGES, 128, 0, stream>>>(X, V, E, eX);

  // 512 tiles per matrix, 8 waves (1 tile each) per 256-thread block.
  gemm_wmma_kernel<false><<<64, 256, 0, stream>>>(X,  W1, hX);
  gemm_wmma_kernel<true ><<<64, 256, 0, stream>>>(eX, W1, hET);

  score_kernel<4><<<NUM_NODES / 4, NUM_EDGES, 0, stream>>>(hX, hET, b1, W2, b2, out);
}
